// GATsSuperGlue_62139586839242
// MI455X (gfx1250) — matrix operand order repaired and checked by reading, not compiled, not run on hardware
//
#include <hip/hip_runtime.h>
#include <hip/hip_bf16.h>
#include <stdint.h>

typedef _Float16 f16;
typedef __attribute__((ext_vector_type(16))) _Float16 v16h;
typedef __attribute__((ext_vector_type(8)))  _Float16 v8h;
typedef __attribute__((ext_vector_type(8)))  float    v8f;

#define TPB 256
#define BM 64
#define BN 128
#define BK 32

// ---------------------------------------------------------------------------
// f32 -> f16 weight conversion (n % 4 == 0 guaranteed at call sites)
// ---------------------------------------------------------------------------
__global__ __launch_bounds__(TPB)
void sg_cvt_f16(const float* __restrict__ src, f16* __restrict__ dst, long n) {
  long i = ((long)blockIdx.x*TPB + threadIdx.x) * 4;
  if (i < n) {
    float4 v = *(const float4*)(src + i);
    dst[i+0] = (f16)v.x; dst[i+1] = (f16)v.y;
    dst[i+2] = (f16)v.z; dst[i+3] = (f16)v.w;
  }
}

// ---------------------------------------------------------------------------
// GEMM: C[b] = scale*(A x B[b]) (+bias[m]) (+residual[b]); f16 WMMA compute,
// f32 accumulate (v_wmma_f32_16x16x32_f16). Software-pipelined with
// double-buffered LDS; A panel staged via global_load_async_to_lds_b128
// (ASYNCcnt) when aMode==0. All four WMMAs issue back-to-back after a
// batched fragment load (keeps the matrix pipe busy, per ISA 7.5).
//   aMode 0: A16 f16 weights [M,K]   (async LDS copy)
//   aMode 1: A16 f16 stored  [K,M]   (gatW transpose)
//   aMode 2: A32 f32 stored  [K,M]   (scores GEMM: mq)
// B: f32 [K,N] (ldb), batched.  M%64==0, N%128==0, K%32==0 at all call sites.
// ---------------------------------------------------------------------------
__global__ __launch_bounds__(TPB)
void sg_gemm_wmma(const f16* __restrict__ A16, const float* __restrict__ A32,
                  long aBatch, int lda, int aMode,
                  const float* __restrict__ B, long bBatch, int ldb,
                  float* __restrict__ C, long cBatch, int ldc,
                  const float* __restrict__ bias,
                  const float* __restrict__ residual, long rBatch,
                  float scale, int M, int N, int K)
{
  __shared__ __align__(16) f16 As[2][BM][BK];
  __shared__ __align__(16) f16 Bs[2][BN][BK];   // transposed: [n][k]
  const int bz = blockIdx.z;
  const int m0 = blockIdx.y * BM;
  const int n0 = blockIdx.x * BN;
  const float* Bb = B + (long)bz * bBatch;
  float* Cb = C + (long)bz * cBatch;
  const int t    = threadIdx.x;
  const int wave = t >> 5;
  const int lane = t & 31;
  const int mt   = wave & 3;          // m-tile of this wave (BM/16 = 4)
  const int ng   = (wave >> 2) << 2;  // first n-tile of this wave (0 or 4)
  const int hs   = (lane >> 4) & 1;   // lane half-select
  const int lm   = lane & 15;

  // ---- B staging: thread owns column n = t&127, k = (t>>7)*16 + i ----
  const int bn = t & 127, bkh = (t >> 7) & 1;
  const float* bptr = Bb + (long)(bkh*16)*ldb + (n0 + bn);

  // ---- A staging coords ----
  // aMode 0 (async): thread copies one 16B chunk; 256 threads = whole 4KB tile
  const int am = t >> 2, akq = t & 3;
  const unsigned aLds0 = (unsigned)(uintptr_t)&As[0][am][akq*8];
  const unsigned aLds1 = (unsigned)(uintptr_t)&As[1][am][akq*8];
  const f16* ap16 = A16 + (long)bz*aBatch + (long)(m0+am)*lda + akq*8;
  // aMode 1/2 (trans): thread owns k = t&31, 8 consecutive m
  const int tk = t & 31, tmo = (t >> 5) * 8;
  const f16*   ap16t = A16 + (long)bz*aBatch + (long)tk*lda + m0 + tmo;
  const float* ap32t = A32 + (long)bz*aBatch + (long)tk*lda + m0 + tmo;

  v8f acc[4] = {};
  v8h  areg;            // aMode 1 staging register
  float4 ar0, ar1;      // aMode 2 staging registers

  // ---------------- prologue: stage tile 0 into buffer 0 ----------------
  {
    float rb[16];
    { const float* p = bptr;
      #pragma unroll
      for (int i = 0; i < 16; ++i) { rb[i] = *p; p += ldb; } }
    bptr += (long)BK * ldb;
    if (aMode == 0) {
      asm volatile("global_load_async_to_lds_b128 %0, %1, off"
                   :: "v"(aLds0), "v"(ap16) : "memory");
      ap16 += BK;
    } else if (aMode == 1) {
      v8h av = *(const v8h*)ap16t; ap16t += (long)BK * lda;
      #pragma unroll
      for (int i = 0; i < 8; ++i) As[0][tmo+i][tk] = av[i];
    } else {
      float4 a0 = *(const float4*)ap32t;
      float4 a1 = *(const float4*)(ap32t + 4);
      ap32t += (long)BK * lda;
      As[0][tmo+0][tk] = (f16)a0.x; As[0][tmo+1][tk] = (f16)a0.y;
      As[0][tmo+2][tk] = (f16)a0.z; As[0][tmo+3][tk] = (f16)a0.w;
      As[0][tmo+4][tk] = (f16)a1.x; As[0][tmo+5][tk] = (f16)a1.y;
      As[0][tmo+6][tk] = (f16)a1.z; As[0][tmo+7][tk] = (f16)a1.w;
    }
    v8h h0, h1;
    #pragma unroll
    for (int i = 0; i < 8; ++i) { h0[i] = (f16)rb[i]; h1[i] = (f16)rb[i+8]; }
    f16* bd = &Bs[0][bn][bkh*16];
    *(v8h*)bd       = h0;
    *(v8h*)(bd + 8) = h1;
    if (aMode == 0)
      asm volatile("s_wait_asynccnt 0x0" ::: "memory");
    __syncthreads();
  }

  const int nk = K / BK;
  #pragma unroll 2
  for (int kt = 0; kt < nk; ++kt) {
    const int cu = kt & 1;
    const int nx = cu ^ 1;
    const bool more = (kt + 1 < nk);
    float rb[16];
    // ---- issue next panel's global traffic BEFORE compute ----
    if (more) {
      { const float* p = bptr;
        #pragma unroll
        for (int i = 0; i < 16; ++i) { rb[i] = *p; p += ldb; } }
      bptr += (long)BK * ldb;
      if (aMode == 0) {
        asm volatile("global_load_async_to_lds_b128 %0, %1, off"
                     :: "v"(cu ? aLds0 : aLds1), "v"(ap16) : "memory");
        ap16 += BK;
      } else if (aMode == 1) {
        areg = *(const v8h*)ap16t; ap16t += (long)BK * lda;
      } else {
        ar0 = *(const float4*)ap32t;
        ar1 = *(const float4*)(ap32t + 4);
        ap32t += (long)BK * lda;
      }
    }
    // ---- batched fragment loads from buffer cu ----
    v16h a;
    {
      // A frag: lanes 0-15 rows M, K {0..7}u{16..23}; lanes 16-31 K+8.
      const f16* ap = &As[cu][mt*16 + lm][hs*8];
      v8h lo = *(const v8h*)ap;
      v8h hi = *(const v8h*)(ap + 16);
      #pragma unroll
      for (int i = 0; i < 8; ++i) { a[i] = lo[i]; a[i+8] = hi[i]; }
    }
    v16h bf[4];
    #pragma unroll
    for (int j = 0; j < 4; ++j) {
      const f16* bp = &Bs[cu][(ng + j)*16 + lm][hs*8];
      v8h b0 = *(const v8h*)bp;
      v8h b1 = *(const v8h*)(bp + 16);
      #pragma unroll
      for (int i = 0; i < 8; ++i) { bf[j][i] = b0[i]; bf[j][i+8] = b1[i]; }
    }
    // ---- four WMMAs back-to-back ----
    #pragma unroll
    for (int j = 0; j < 4; ++j)
      acc[j] = __builtin_amdgcn_wmma_f32_16x16x32_f16(
                 false, a, false, bf[j], (short)0, acc[j], false, false);
    // ---- finish staging buffer nx AFTER compute (loads have landed) ----
    if (more) {
      if (aMode == 1) {
        #pragma unroll
        for (int i = 0; i < 8; ++i) As[nx][tmo+i][tk] = areg[i];
      } else if (aMode == 2) {
        As[nx][tmo+0][tk] = (f16)ar0.x; As[nx][tmo+1][tk] = (f16)ar0.y;
        As[nx][tmo+2][tk] = (f16)ar0.z; As[nx][tmo+3][tk] = (f16)ar0.w;
        As[nx][tmo+4][tk] = (f16)ar1.x; As[nx][tmo+5][tk] = (f16)ar1.y;
        As[nx][tmo+6][tk] = (f16)ar1.z; As[nx][tmo+7][tk] = (f16)ar1.w;
      }
      v8h h0, h1;
      #pragma unroll
      for (int i = 0; i < 8; ++i) { h0[i] = (f16)rb[i]; h1[i] = (f16)rb[i+8]; }
      f16* bd = &Bs[nx][bn][bkh*16];
      *(v8h*)bd       = h0;
      *(v8h*)(bd + 8) = h1;
      if (aMode == 0)
        asm volatile("s_wait_asynccnt 0x0" ::: "memory");
    }
    __syncthreads();
  }

  #pragma unroll
  for (int j = 0; j < 4; ++j) {
    int n = n0 + (ng + j)*16 + lm;
    #pragma unroll
    for (int f = 0; f < 8; ++f) {
      int m = m0 + mt*16 + f + hs*8;     // C/D layout: VGPR f -> M=f / f+8
      float v = acc[j][f] * scale;
      if (bias) v += bias[m];
      long off = (long)m*ldc + n;
      if (residual) v += residual[(long)bz*rBatch + off];
      Cb[off] = v;
    }
  }
}

// ---------------- elementwise / reduction helpers --------------------------
__global__ __launch_bounds__(TPB)
void sg_elu1(float* __restrict__ x, long n) {
  long i = (long)blockIdx.x*TPB + threadIdx.x;
  if (i < n) { float v = x[i]; x[i] = v > 0.f ? v + 1.f : __expf(v); }
}

__global__ __launch_bounds__(TPB)
void sg_copy_batched(const float* __restrict__ src, long sB,
                     float* __restrict__ dst, long dB, long per, int nb) {
  long i = (long)blockIdx.x*TPB + threadIdx.x;
  if (i < per*nb) { long b = i/per, r = i%per; dst[b*dB + r] = src[b*sB + r]; }
}

__global__ __launch_bounds__(TPB)
void sg_rowsum(const float* __restrict__ X, float* __restrict__ out, int N) {
  long base = (long)blockIdx.x * N;
  __shared__ float sh[TPB];
  float s = 0.f;
  for (int i = threadIdx.x; i < N; i += TPB) s += X[base+i];
  sh[threadIdx.x] = s; __syncthreads();
  for (int o = TPB/2; o > 0; o >>= 1) {
    if (threadIdx.x < o) sh[threadIdx.x] += sh[threadIdx.x+o];
    __syncthreads();
  }
  if (threadIdx.x == 0) out[blockIdx.x] = sh[0];
}

__global__ __launch_bounds__(TPB)
void sg_instnorm_relu(float* __restrict__ x, int N) {
  long base = (long)blockIdx.x * N;
  __shared__ float ss[TPB], sq[TPB];
  float s = 0.f, q = 0.f;
  for (int i = threadIdx.x; i < N; i += TPB) { float v = x[base+i]; s += v; q += v*v; }
  ss[threadIdx.x] = s; sq[threadIdx.x] = q; __syncthreads();
  for (int o = TPB/2; o > 0; o >>= 1) {
    if (threadIdx.x < o) { ss[threadIdx.x] += ss[threadIdx.x+o]; sq[threadIdx.x] += sq[threadIdx.x+o]; }
    __syncthreads();
  }
  float mean = ss[0] / N;
  float var  = sq[0] / N - mean*mean;           // biased, matches jnp.var
  float inv  = rsqrtf(var + 1e-5f);
  for (int i = threadIdx.x; i < N; i += TPB) {
    float v = (x[base+i] - mean) * inv;
    x[base+i] = v > 0.f ? v : 0.f;
  }
}

// KV[b,h][q][d] = sum_m V[(q*4+h), m] * K[(d*4+h), m]   (v/m and *m cancel)
__global__ __launch_bounds__(TPB)
void sg_kv(const float* __restrict__ K, const float* __restrict__ V,
           float* __restrict__ KV, int N) {
  int bh = blockIdx.x, b = bh >> 2, h = bh & 3;
  const float* Kb = K + (long)b*256*N;
  const float* Vb = V + (long)b*256*N;
  __shared__ float Kt[64][64], Vt[64][64];
  int t = threadIdx.x, q = t & 63, dg = t >> 6;   // dg in 0..3
  float acc[16] = {};
  for (int mc = 0; mc < N; mc += 64) {
    #pragma unroll
    for (int i = 0; i < 16; ++i) {
      int idx = t + i*TPB, d = idx >> 6, m = idx & 63;
      Kt[d][m] = Kb[(long)(d*4+h)*N + mc + m];
      Vt[d][m] = Vb[(long)(d*4+h)*N + mc + m];
    }
    __syncthreads();
    for (int m = 0; m < 64; ++m) {
      float vq = Vt[q][m];
      #pragma unroll
      for (int j = 0; j < 16; ++j) acc[j] += vq * Kt[dg*16+j][m];
    }
    __syncthreads();
  }
  float* o = KV + (long)bh*4096 + (long)q*64 + dg*16;
  #pragma unroll
  for (int j = 0; j < 16; ++j) o[j] = acc[j];
}

// O[b, q*4+h, m] = Z[h,m] * sum_d Q[d*4+h, m] * KV[b,h,q,d]
__global__ __launch_bounds__(TPB)
void sg_linattn_out(const float* __restrict__ Q, const float* __restrict__ ksum,
                    const float* __restrict__ KV, float* __restrict__ O, int N) {
  int bx = blockIdx.x, b = bx / N, m = bx % N;
  __shared__ float qv[256];
  __shared__ float z[4];
  int t = threadIdx.x;
  qv[t] = Q[(long)b*256*N + (long)t*N + m];
  __syncthreads();
  if (t < 4) {
    float s = 0.f;
    for (int d = 0; d < 64; ++d) s += qv[d*4+t] * ksum[b*256 + d*4+t];
    z[t] = 1.0f / (s + 1e-6f);
  }
  __syncthreads();
  int q = t >> 2, h = t & 3;
  const float* kvp = KV + (long)(b*4+h)*4096 + (long)q*64;
  float acc = 0.f;
  #pragma unroll 8
  for (int d = 0; d < 64; ++d) acc += qv[d*4+h] * kvp[d];
  O[(long)b*256*N + (long)(q*4+h)*N + m] = acc * z[h];
}

// out[b,c] = sum_d a[d] * Y[b, d, c]
__global__ __launch_bounds__(TPB)
void sg_coldot(const float* __restrict__ Y, const float* __restrict__ a,
               float* __restrict__ out, long cols, int nb) {
  long i = (long)blockIdx.x*TPB + threadIdx.x;
  if (i >= cols*nb) return;
  long b = i / cols, c = i % cols;
  const float* Yb = Y + b*256*cols + c;
  float s = 0.f;
  for (int d = 0; d < 256; ++d) s += a[d] * Yb[(long)d*cols];
  out[i] = s;
}

__global__ __launch_bounds__(TPB)
void sg_gat_combine(const float* __restrict__ Y3, const float* __restrict__ Y2,
                    const float* __restrict__ t3, const float* __restrict__ s3,
                    const float* __restrict__ s2, float* __restrict__ out,
                    int n, int L) {
  int bx = blockIdx.x, b = bx / n, nn = bx % n;
  long NL = (long)n * L;
  __shared__ float att[9];
  if (threadIdx.x == 0) {
    float t3v = t3[bx], e[9];
    float x0 = t3v + s3[bx]; e[0] = x0 > 0.f ? x0 : 0.2f*x0;
    for (int l = 1; l <= L; ++l) {
      float x = t3v + s2[(long)b*NL + (long)nn*L + (l-1)];
      e[l] = x > 0.f ? x : 0.2f*x;
    }
    float mx = e[0];
    for (int l = 1; l <= L; ++l) mx = fmaxf(mx, e[l]);
    float s = 0.f;
    for (int l = 0; l <= L; ++l) { e[l] = __expf(e[l]-mx); s += e[l]; }
    for (int l = 0; l <= L; ++l) att[l] = e[l] / s;
  }
  __syncthreads();
  int d = threadIdx.x;
  float h = att[0] * Y3[(long)b*256*n + (long)d*n + nn];
  for (int l = 1; l <= L; ++l)
    h += att[l] * Y2[(long)b*256*NL + (long)d*NL + (long)nn*L + (l-1)];
  out[(long)b*256*n + (long)d*n + nn] = h > 0.f ? h : __expf(h) - 1.f;  // elu
}

__global__ __launch_bounds__(TPB)
void sg_l2norm_cols(float* __restrict__ X, int N, int nb) {
  long i = (long)blockIdx.x*TPB + threadIdx.x;
  if (i >= (long)N*nb) return;
  long b = i / N, c = i % N;
  float* p = X + b*256*(long)N + c;
  float s = 0.f;
  for (int d = 0; d < 256; ++d) { float v = p[(long)d*N]; s += v*v; }
  float inv = 1.0f / fmaxf(sqrtf(s), 1e-12f);
  for (int d = 0; d < 256; ++d) p[(long)d*N] *= inv;
}

__global__ __launch_bounds__(TPB)
void sg_row_softmax_stats(const float* __restrict__ X, float* __restrict__ rmax,
                          float* __restrict__ rsum, int N) {
  long base = (long)blockIdx.x * N;
  __shared__ float sh[TPB];
  float mx = -INFINITY;
  for (int i = threadIdx.x; i < N; i += TPB) mx = fmaxf(mx, X[base+i]);
  sh[threadIdx.x] = mx; __syncthreads();
  for (int o = TPB/2; o > 0; o >>= 1) {
    if (threadIdx.x < o) sh[threadIdx.x] = fmaxf(sh[threadIdx.x], sh[threadIdx.x+o]);
    __syncthreads();
  }
  mx = sh[0]; __syncthreads();
  float s = 0.f;
  for (int i = threadIdx.x; i < N; i += TPB) s += __expf(X[base+i] - mx);
  sh[threadIdx.x] = s; __syncthreads();
  for (int o = TPB/2; o > 0; o >>= 1) {
    if (threadIdx.x < o) sh[threadIdx.x] += sh[threadIdx.x+o];
    __syncthreads();
  }
  if (threadIdx.x == 0) { rmax[blockIdx.x] = mx; rsum[blockIdx.x] = sh[0]; }
}

__global__ __launch_bounds__(TPB)
void sg_col_softmax_stats(const float* __restrict__ X, float* __restrict__ cmax,
                          float* __restrict__ csum, int Nr, int Nc, int nb) {
  long i = (long)blockIdx.x*TPB + threadIdx.x;
  if (i >= (long)nb*Nc) return;
  long b = i / Nc; int m = (int)(i % Nc);
  const float* Xb = X + b*(long)Nr*Nc + m;
  float mx = -INFINITY;
  for (int n = 0; n < Nr; ++n) mx = fmaxf(mx, Xb[(long)n*Nc]);
  float s = 0.f;
  for (int n = 0; n < Nr; ++n) s += __expf(Xb[(long)n*Nc] - mx);
  cmax[i] = mx; csum[i] = s;
}

__global__ __launch_bounds__(TPB)
void sg_conf(const float* __restrict__ S, const float* __restrict__ rmax,
             const float* __restrict__ rsum, const float* __restrict__ cmax,
             const float* __restrict__ csum, float* __restrict__ conf,
             int Nr, int Nc, long total) {
  long i = (long)blockIdx.x*TPB + threadIdx.x;
  if (i >= total) return;
  long per = (long)Nr*Nc, b = i / per, r2 = i % per;
  int r = (int)(r2 / Nc), c = (int)(r2 % Nc);
  float s = S[i];
  long rb = b*Nr + r, cb = b*Nc + c;
  conf[i] = (__expf(s - rmax[rb]) / rsum[rb]) * (__expf(s - cmax[cb]) / csum[cb]);
}

__global__ __launch_bounds__(TPB)
void sg_argmax_rows(const float* __restrict__ X, int* __restrict__ idx,
                    float* __restrict__ val, int N) {
  long base = (long)blockIdx.x * N;
  float bm = -INFINITY; int bi = 0;
  for (int i = threadIdx.x; i < N; i += TPB) {
    float v = X[base+i];
    if (v > bm) { bm = v; bi = i; }
  }
  __shared__ float sv[TPB]; __shared__ int si[TPB];
  sv[threadIdx.x] = bm; si[threadIdx.x] = bi; __syncthreads();
  for (int o = TPB/2; o > 0; o >>= 1) {
    if (threadIdx.x < o) {
      float v2 = sv[threadIdx.x+o]; int i2 = si[threadIdx.x+o];
      if (v2 > sv[threadIdx.x] || (v2 == sv[threadIdx.x] && i2 < si[threadIdx.x])) {
        sv[threadIdx.x] = v2; si[threadIdx.x] = i2;
      }
    }
    __syncthreads();
  }
  if (threadIdx.x == 0) { idx[blockIdx.x] = si[0]; val[blockIdx.x] = sv[0]; }
}

__global__ __launch_bounds__(TPB)
void sg_argmax_cols(const float* __restrict__ X, int* __restrict__ idx,
                    float* __restrict__ val, int Nr, int Nc, int nb) {
  long i = (long)blockIdx.x*TPB + threadIdx.x;
  if (i >= (long)nb*Nc) return;
  long b = i / Nc; int m = (int)(i % Nc);
  const float* Xb = X + b*(long)Nr*Nc + m;
  float bm = -INFINITY; int bi = 0;
  for (int n = 0; n < Nr; ++n) { float v = Xb[(long)n*Nc]; if (v > bm) { bm = v; bi = n; } }
  idx[i] = bi; val[i] = bm;
}

// Mutual matching, batch 0. Indices written as float (d_out is a float buffer).
__global__ __launch_bounds__(TPB)
void sg_match0(const int* __restrict__ i0, const float* __restrict__ m0,
               const int* __restrict__ i1, float* __restrict__ ms0,
               float* __restrict__ ind0, int* __restrict__ valid0, int n) {
  int nn = blockIdx.x*TPB + threadIdx.x;
  if (nn >= n) return;
  int j = i0[nn];
  bool mut = (i1[j] == nn);
  float ms = mut ? m0[nn] : 0.f;
  ms0[nn] = ms;
  bool val = mut && (ms > 0.2f);
  valid0[nn] = val ? 1 : 0;
  ind0[nn] = val ? (float)j : -1.0f;
}

__global__ __launch_bounds__(TPB)
void sg_match1(const int* __restrict__ i0, const int* __restrict__ i1,
               const float* __restrict__ ms0, const int* __restrict__ valid0,
               float* __restrict__ ms1, float* __restrict__ ind1, int n) {
  int m = blockIdx.x*TPB + threadIdx.x;
  if (m >= n) return;
  int j = i1[m];
  bool mut = (i0[j] == m);
  ms1[m] = mut ? ms0[j] : 0.f;
  bool val = mut && (valid0[j] != 0);
  ind1[m] = val ? (float)j : -1.0f;
}

// ---------------------------------------------------------------------------
extern "C" void kernel_launch(void* const* d_in, const int* in_sizes, int n_in,
                              void* d_out, int out_size, void* d_ws, size_t ws_size,
                              hipStream_t stream)
{
  (void)in_sizes; (void)n_in; (void)out_size; (void)ws_size;
  const int Dd = 256, Nq = 3072, NB = 2, L8 = 8;
  const long NL = (long)Nq * L8;
  const long DN = (long)Dd * Nq;

  const float* in_dq  = (const float*)d_in[0];
  const float* in_d3  = (const float*)d_in[1];
  const float* d2db   = (const float*)d_in[2];
  const float* projw  = (const float*)d_in[3];
  const float* projb  = (const float*)d_in[4];
  const float* mergew = (const float*)d_in[5];
  const float* mergeb = (const float*)d_in[6];
  const float* mlp1w  = (const float*)d_in[7];
  const float* mlp1b  = (const float*)d_in[8];
  const float* mlp2w  = (const float*)d_in[9];
  const float* mlp2b  = (const float*)d_in[10];
  const float* gatW   = (const float*)d_in[11];
  const float* gata   = (const float*)d_in[12];
  const float* finw   = (const float*)d_in[13];
  const float* finb   = (const float*)d_in[14];

  // ---- workspace carve-up (phase-exclusive aliases noted) ----
  float* wsf = (float*)d_ws;
  size_t cur = 0;
  auto alloc  = [&](long nel) { float* p = wsf + cur; cur += (size_t)nel; return p; };
  auto allocH = [&](long nel) { f16* p = (f16*)(wsf + cur); cur += (size_t)((nel+1)/2); return p; };

  // f16 weight arena
  const long nProjw = 8L*3*Dd*Dd, nMergew = 8L*Dd*Dd, nMlp1w = 8L*512*512;
  const long nMlp2w = 8L*Dd*512,  nGatW   = 4L*Dd*Dd, nFinw  = (long)Dd*Dd;
  f16* projwH  = allocH(nProjw);
  f16* mergewH = allocH(nMergew);
  f16* mlp1wH  = allocH(nMlp1w);
  f16* mlp2wH  = allocH(nMlp2w);
  f16* gatWH   = allocH(nGatW);
  f16* finwH   = allocH(nFinw);

  float* dqA = alloc(NB*DN); float* dqB = alloc(NB*DN);
  float* d3A = alloc(NB*DN); float* d3B = alloc(NB*DN);
  float* Qb  = alloc(NB*DN); float* Kb = alloc(NB*DN);
  float* Vb  = alloc(NB*DN); float* Ob = alloc(NB*DN);
  float* ksum = alloc(NB*Dd);
  float* KVb  = alloc((long)NB*4*64*64);
  float* big  = alloc((long)NB*Nq*Nq);      // 18.9M floats: largest phase user
  float* Ybuf = big;                        // attn: [b,2D,N]
  float* Tbuf = big + NB*2*DN;              // attn: [b,2D,N]
  float* Y2   = big;                        // GAT:  [b,D,N*L]
  float* Sbuf = big;                        // final: scores [b,N,N]
  float* Y3   = Ob;                         // GAT (alias of attn O)
  float* s2 = alloc(NB*NL); float* s3 = alloc(NB*Nq); float* t3 = alloc(NB*Nq);
  float* MQ = Qb; float* MD = Kb;           // final (alias of attn Q/K)
  float* rmax = alloc(NB*Nq); float* rsum = alloc(NB*Nq);
  float* cmax = alloc(NB*Nq); float* csum = alloc(NB*Nq);
  float* m0v  = alloc(NB*Nq); float* m1v  = alloc(NB*Nq);
  int* i0 = (int*)alloc(NB*Nq);
  int* i1 = (int*)alloc(NB*Nq);
  int* valid0 = (int*)alloc(Nq);

  // ---- pre-convert all weights to f16 (once per call; all sizes %1024==0)
  auto cvt = [&](const float* s, f16* d, long n) {
    sg_cvt_f16<<<(int)(n/(TPB*4)), TPB, 0, stream>>>(s, d, n);
  };
  cvt(projw,  projwH,  nProjw);
  cvt(mergew, mergewH, nMergew);
  cvt(mlp1w,  mlp1wH,  nMlp1w);
  cvt(mlp2w,  mlp2wH,  nMlp2w);
  cvt(gatW,   gatWH,   nGatW);
  cvt(finw,   finwH,   nFinw);

  hipMemcpyAsync(dqA, in_dq, sizeof(float)*NB*DN, hipMemcpyDeviceToDevice, stream);
  hipMemcpyAsync(d3A, in_d3, sizeof(float)*NB*DN, hipMemcpyDeviceToDevice, stream);

  auto gemm = [&](const f16* A16, const float* A32, long aB, int lda, int aMode,
                  const float* B, long bB, int ldb,
                  float* C, long cB, int ldc,
                  const float* bias, const float* res, long rB,
                  float scale, int M, int Nn, int K) {
    dim3 grid(Nn/BN, M/BM, NB);
    sg_gemm_wmma<<<grid, TPB, 0, stream>>>(A16, A32, aB, lda, aMode, B, bB, ldb,
                                           C, cB, ldc, bias, res, rB,
                                           scale, M, Nn, K);
  };

  float *cur_dq = dqA, *alt_dq = dqB, *cur_d3 = d3A, *alt_d3 = d3B;
  int ai = 0, gi = 0;

  auto attn_update = [&](const float* x, float* xout, const float* src) {
    const f16*   pw = projwH + (long)ai*3*Dd*Dd;
    const float* pb = projb  + (long)ai*3*Dd;
    gemm(pw,           nullptr, 0, Dd, 0, x,   DN, Nq, Qb, DN, Nq, pb,        nullptr, 0, 1.f, Dd, Nq, Dd);
    gemm(pw +   Dd*Dd, nullptr, 0, Dd, 0, src, DN, Nq, Kb, DN, Nq, pb + Dd,   nullptr, 0, 1.f, Dd, Nq, Dd);
    gemm(pw + 2*Dd*Dd, nullptr, 0, Dd, 0, src, DN, Nq, Vb, DN, Nq, pb + 2*Dd, nullptr, 0, 1.f, Dd, Nq, Dd);
    sg_elu1<<<(int)((NB*DN)/TPB), TPB, 0, stream>>>(Qb, NB*DN);
    sg_elu1<<<(int)((NB*DN)/TPB), TPB, 0, stream>>>(Kb, NB*DN);
    sg_rowsum<<<NB*Dd, TPB, 0, stream>>>(Kb, ksum, Nq);
    sg_kv<<<NB*4, TPB, 0, stream>>>(Kb, Vb, KVb, Nq);
    sg_linattn_out<<<NB*Nq, TPB, 0, stream>>>(Qb, ksum, KVb, Ob, Nq);
    // merge -> lower half of Ybuf ; copy x -> upper half  (implicit concat)
    gemm(mergewH + (long)ai*Dd*Dd, nullptr, 0, Dd, 0, Ob, DN, Nq, Ybuf + DN, 2*DN, Nq,
         mergeb + (long)ai*Dd, nullptr, 0, 1.f, Dd, Nq, Dd);
    sg_copy_batched<<<(int)((NB*DN)/TPB), TPB, 0, stream>>>(x, DN, Ybuf, 2*DN, DN, NB);
    gemm(mlp1wH + (long)ai*512*512, nullptr, 0, 512, 0, Ybuf, 2*DN, Nq, Tbuf, 2*DN, Nq,
         mlp1b + (long)ai*512, nullptr, 0, 1.f, 512, Nq, 512);
    sg_instnorm_relu<<<NB*512, TPB, 0, stream>>>(Tbuf, Nq);
    gemm(mlp2wH + (long)ai*Dd*512, nullptr, 0, 512, 0, Tbuf, 2*DN, Nq, xout, DN, Nq,
         mlp2b + (long)ai*Dd, x, DN, 1.f, Dd, Nq, 512);
  };

  for (int rep = 0; rep < 4; ++rep) {
    // ---- GATs ----
    const f16*   Wg = gatWH + (long)gi*Dd*Dd;
    const float* ag = gata  + (long)gi*2*Dd;
    gemm(Wg, nullptr, 0, Dd, 1, cur_d3, DN, Nq, Y3, DN, Nq, nullptr, nullptr, 0, 1.f, Dd, Nq, Dd);
    gemm(Wg, nullptr, 0, Dd, 1, d2db, (long)Dd*NL, (int)NL, Y2, (long)Dd*NL, (int)NL,
         nullptr, nullptr, 0, 1.f, Dd, (int)NL, Dd);
    sg_coldot<<<(int)((NB*NL)/TPB), TPB, 0, stream>>>(Y2, ag + Dd, s2, NL, NB);
    sg_coldot<<<(NB*Nq)/TPB, TPB, 0, stream>>>(Y3, ag + Dd, s3, Nq, NB);
    sg_coldot<<<(NB*Nq)/TPB, TPB, 0, stream>>>(Y3, ag,      t3, Nq, NB);
    sg_gat_combine<<<NB*Nq, TPB, 0, stream>>>(Y3, Y2, t3, s3, s2, alt_d3, Nq, L8);
    { float* tp = cur_d3; cur_d3 = alt_d3; alt_d3 = tp; }
    gi++;
    // ---- self ----
    attn_update(cur_dq, alt_dq, cur_dq);
    attn_update(cur_d3, alt_d3, cur_d3);
    { float* tp = cur_dq; cur_dq = alt_dq; alt_dq = tp; }
    { float* tp = cur_d3; cur_d3 = alt_d3; alt_d3 = tp; }
    ai++;
    // ---- cross (d3's source must be the OLD dq -> double buffering) ----
    attn_update(cur_dq, alt_dq, cur_d3);
    attn_update(cur_d3, alt_d3, cur_dq);
    { float* tp = cur_dq; cur_dq = alt_dq; alt_dq = tp; }
    { float* tp = cur_d3; cur_d3 = alt_d3; alt_d3 = tp; }
    ai++;
  }

  // ---- final projection, normalize, scores, dual softmax, matching ----
  gemm(finwH, nullptr, 0, Dd, 0, cur_dq, DN, Nq, MQ, DN, Nq, finb, nullptr, 0, 1.f, Dd, Nq, Dd);
  gemm(finwH, nullptr, 0, Dd, 0, cur_d3, DN, Nq, MD, DN, Nq, finb, nullptr, 0, 1.f, Dd, Nq, Dd);
  sg_l2norm_cols<<<(NB*Nq)/TPB, TPB, 0, stream>>>(MQ, Nq, NB);
  sg_l2norm_cols<<<(NB*Nq)/TPB, TPB, 0, stream>>>(MD, Nq, NB);
  // scores[n,m] = (1/SCALE) * sum_d mq[d,n]*md[d,m]  -> aMode 2 (f32 trans A)
  gemm(nullptr, MQ, DN, Nq, 2, MD, DN, Nq, Sbuf, (long)Nq*Nq, Nq,
       nullptr, nullptr, 0, 10.0f, Nq, Nq, Dd);
  sg_row_softmax_stats<<<NB*Nq, TPB, 0, stream>>>(Sbuf, rmax, rsum, Nq);
  sg_col_softmax_stats<<<(NB*Nq)/TPB, TPB, 0, stream>>>(Sbuf, cmax, csum, Nq, Nq, NB);

  float* conf = (float*)d_out + 4*Nq;
  long confTotal = (long)NB*Nq*Nq;
  sg_conf<<<(int)(confTotal/TPB), TPB, 0, stream>>>(Sbuf, rmax, rsum, cmax, csum,
                                                    conf, Nq, Nq, confTotal);
  sg_argmax_rows<<<NB*Nq, TPB, 0, stream>>>(conf, i0, m0v, Nq);
  sg_argmax_cols<<<(NB*Nq)/TPB, TPB, 0, stream>>>(conf, i1, m1v, Nq, Nq, NB);

  float* ind0 = (float*)d_out;
  float* ind1 = ind0 + Nq;
  float* ms0  = (float*)d_out + 2*Nq;
  float* ms1  = (float*)d_out + 3*Nq;
  sg_match0<<<Nq/TPB, TPB, 0, stream>>>(i0, m0v, i1, ms0, ind0, valid0, Nq);
  sg_match1<<<Nq/TPB, TPB, 0, stream>>>(i0, i1, ms0, valid0, ms1, ind1, Nq);
}
